// QuantizedLinear_283467842173
// MI455X (gfx1250) — compile-verified, matrix operand order checked by reading, not run on hardware
//
#include <hip/hip_runtime.h>

typedef __attribute__((ext_vector_type(16))) _Float16 v16h;
typedef __attribute__((ext_vector_type(8)))  float    v8f;
typedef __attribute__((ext_vector_type(4)))  float    v4f;
typedef __attribute__((ext_vector_type(4)))  int      v4i;
typedef __attribute__((ext_vector_type(4)))  _Float16 v4h;

// pointer flavors for the async-copy builtin (probe-verified signature:
// (int4 AS1*, int4 AS3*, imm offset, imm cpol))
typedef __attribute__((address_space(1))) v4i gv4i;
typedef __attribute__((address_space(3))) v4i lv4i;

#define M_TOK 8192    // B*S = 4*2048
#define K_DIM 4096
#define N_DIM 16384

#define BM 256        // block tile rows   (M/BM = 32 passes over the weight)
#define BN 64         // block tile cols
#define BK 32         // K step (one WMMA K)
#define NSTAGE 3      // circular LDS pipeline depth -> single barrier/iter

// async copies issued per thread per K-step: A = BM*BK*2B/16B/256thr = 4, W = 1
#define CP_PER_ITER 5

// ---------------- async global->LDS (CDNA5), with safe fallback ----------------
#if __has_builtin(__builtin_amdgcn_global_load_async_to_lds_b128)
#define ASYNC_LDS 1
#endif

__device__ __forceinline__ void copy16_to_lds(const _Float16* g, _Float16* l) {
#ifdef ASYNC_LDS
    __builtin_amdgcn_global_load_async_to_lds_b128((gv4i*)g, (lv4i*)l, 0, 0);
#else
    *(v4i*)l = *(const v4i*)g;
#endif
}

// wait until only the newest batch (CP_PER_ITER copies) is still in flight
__device__ __forceinline__ void async_wait_keep_newest() {
#ifdef ASYNC_LDS
#if __has_builtin(__builtin_amdgcn_s_wait_asynccnt)
    __builtin_amdgcn_s_wait_asynccnt(CP_PER_ITER);
#else
    asm volatile("s_wait_asynccnt %0" :: "i"(CP_PER_ITER) : "memory");
#endif
#endif
}

__device__ __forceinline__ void async_wait_all() {
#ifdef ASYNC_LDS
#if __has_builtin(__builtin_amdgcn_s_wait_asynccnt)
    __builtin_amdgcn_s_wait_asynccnt(0);
#else
    asm volatile("s_wait_asynccnt 0" ::: "memory");
#endif
#endif
}

// ---------------- precision-conversion prepass kernels ----------------
// one-shot streaming reads: nontemporal, so they don't evict the f16 weight
// image we want resident in the 192MB L2 during the GEMM.
__global__ void cvt_act_f16(const float* __restrict__ in, _Float16* __restrict__ out) {
    size_t i = (size_t)blockIdx.x * blockDim.x + threadIdx.x;
    v4f a = __builtin_nontemporal_load(&((const v4f*)in)[i]);
    v4h h;
    h.x = (_Float16)a.x; h.y = (_Float16)a.y;
    h.z = (_Float16)a.z; h.w = (_Float16)a.w;
    ((v4h*)out)[i] = h;   // regular store: A16 is reused by the GEMM
}

__global__ void cvt_w_f16(const int* __restrict__ in, _Float16* __restrict__ out) {
    size_t i = (size_t)blockIdx.x * blockDim.x + threadIdx.x;
    v4i a = __builtin_nontemporal_load(&((const v4i*)in)[i]);
    v4h h;
    h.x = (_Float16)a.x; h.y = (_Float16)a.y;
    h.z = (_Float16)a.z; h.w = (_Float16)a.w;
    ((v4h*)out)[i] = h;   // regular store: W16 should stay L2-resident
}

// ---------------- WMMA GEMM: out[M,N] = A16[M,K] * W16[N,K]^T * scale[N] + bias[N]
// Block tile 256x64, 8 waves (4x2): wave tile 64x32 = 4x2 WMMA tiles (8 wmma/K-step).
// 3-stage circular LDS pipeline fed by async global->LDS copies; 1 barrier/iter.
__global__ __launch_bounds__(256)
void w8a16_wmma_gemm(const _Float16* __restrict__ A,
                     const _Float16* __restrict__ W,
                     const float* __restrict__ scale,
                     const float* __restrict__ bias,
                     float* __restrict__ out) {
    // 96-byte row stride: every 32B v16h fragment load aligned + bank-spread
    __shared__ __align__(32) _Float16 As[NSTAGE][BM][48];
    __shared__ __align__(32) _Float16 Ws[NSTAGE][BN][48];

    const int t    = threadIdx.x;
    const int lane = t & 31;
    const int wv   = t >> 5;
    const int wm   = (wv & 3) * 64;   // wave row offset in block tile
    const int wn   = (wv >> 2) * 32;  // wave col offset in block tile

    const size_t m0 = (size_t)blockIdx.y * BM;
    const size_t n0 = (size_t)blockIdx.x * BN;

    v8f acc[4][2] = {};

    const int l16   = lane & 15;
    const int khalf = lane >> 4;      // K-half (0..15 vs 16..31) owned by this lane

    // per-thread copy coordinates (16B chunks, BK=32 halves = 4 chunks/row)
    const int crow = t >> 2, coff = (t & 3) * 8;

    auto stage = [&](int k, int b) {
#pragma unroll
        for (int i = 0; i < 4; ++i)
            copy16_to_lds(&A[(m0 + crow + i * 64) * K_DIM + k + coff],
                          &As[b][crow + i * 64][coff]);
        copy16_to_lds(&W[(n0 + crow) * K_DIM + k + coff], &Ws[b][crow][coff]);
    };

    auto compute = [&](int b) {
        v16h af[4], bf[2];
#pragma unroll
        for (int tm = 0; tm < 4; ++tm)
            af[tm] = *(const v16h*)&As[b][wm + tm * 16 + l16][khalf * 16];
#pragma unroll
        for (int tn = 0; tn < 2; ++tn)
            bf[tn] = *(const v16h*)&Ws[b][wn + tn * 16 + l16][khalf * 16];
#pragma unroll
        for (int tm = 0; tm < 4; ++tm)
#pragma unroll
            for (int tn = 0; tn < 2; ++tn)
                acc[tm][tn] = __builtin_amdgcn_wmma_f32_16x16x32_f16(
                    /*neg_a=*/false, af[tm], /*neg_b=*/false, bf[tn],
                    /*c_mod=*/(short)0, acc[tm][tn],
                    /*reuse_a=*/false, /*reuse_b=*/false);
    };

    // ---- prologue: stage K-step 0 into buffer 0
    stage(0, 0);

    // ---- steady state: branch-free; issue (i+1) into buf (i+1)%3, wait so the
    //      older batch is landed, one barrier, compute buf i%3.
    int buf = 0, nxt = 1;
#pragma unroll 1
    for (int k0 = 0; k0 < K_DIM - BK; k0 += BK) {
        stage(k0 + BK, nxt);
        async_wait_keep_newest();
        __syncthreads();          // buf's tile visible to every wave
        compute(buf);
        buf = nxt;
        nxt = (nxt == NSTAGE - 1) ? 0 : nxt + 1;
    }

    // ---- peeled final K-step
    async_wait_all();
    __syncthreads();
    compute(buf);

    // ---- epilogue: D layout -> lane L, VGPR r holds (M = r + 8*(L>>4), N = L&15)
    // Output is written once and never re-read: nontemporal stores keep the
    // f16 weight image resident in L2 across M-row passes.
    const int lhi = lane >> 4;
#pragma unroll
    for (int tn = 0; tn < 2; ++tn) {
        size_t n = n0 + wn + tn * 16 + l16;
        float s = scale[n];
        float b = bias[n];
#pragma unroll
        for (int tm = 0; tm < 4; ++tm) {
            size_t mbase = m0 + wm + tm * 16 + lhi * 8;
#pragma unroll
            for (int r = 0; r < 8; ++r)
                __builtin_nontemporal_store(acc[tm][tn][r] * s + b,
                                            &out[(mbase + r) * N_DIM + n]);
        }
    }
}

extern "C" void kernel_launch(void* const* d_in, const int* in_sizes, int n_in,
                              void* d_out, int out_size, void* d_ws, size_t ws_size,
                              hipStream_t stream) {
    (void)in_sizes; (void)n_in; (void)out_size; (void)ws_size;

    const float* inp   = (const float*)d_in[0];   // [4,2048,4096] fp32
    const int*   qw    = (const int*)d_in[1];     // [16384,4096] int32 (int8-valued)
    const float* sc    = (const float*)d_in[2];   // [16384]
    const float* bs    = (const float*)d_in[3];   // [16384]
    float*       out   = (float*)d_out;           // [4,2048,16384] fp32

    _Float16* A16 = (_Float16*)d_ws;                                        // 64 MiB
    _Float16* W16 = (_Float16*)((char*)d_ws + (size_t)M_TOK * K_DIM * 2);   // 128 MiB

    // prepass: fp32->f16 activations, int32->f16 weights (4 elems/thread)
    cvt_act_f16<<<(M_TOK * (size_t)K_DIM) / 4 / 256, 256, 0, stream>>>(inp, A16);
    cvt_w_f16<<<((size_t)N_DIM * K_DIM) / 4 / 256, 256, 0, stream>>>(qw, W16);

    // blockIdx.x (N) varies fastest: all 256 N-blocks of one M-row share the
    // same 2MB A16 slice in L2 while streaming the 128MB W16 image.
    dim3 grid(N_DIM / BN, M_TOK / BM);
    w8a16_wmma_gemm<<<grid, 256, 0, stream>>>(A16, W16, sc, bs, out);
}
